// Attention_11940009083267
// MI455X (gfx1250) — matrix operand order repaired and checked by reading, not compiled, not run on hardware
//
#include <hip/hip_runtime.h>
#include <float.h>
#include <stdint.h>

#define H_   12
#define D_   64
#define B_   2
#define N_   2048
#define C_   768
#define HD_  768
#define QKV_ 2304

typedef __attribute__((ext_vector_type(16))) __bf16 v16bf;
typedef __attribute__((ext_vector_type(8)))  __bf16 v8bf;
typedef __attribute__((ext_vector_type(8)))  float  v8f;

// ---- WMMA fragment loaders (CDNA5 wave32 16x16x32 bf16 layouts) ----
static __device__ __forceinline__ v16bf frag_a(const __bf16* tile, int ldk, int k0, int lane) {
  const __bf16* p = tile + (size_t)(lane & 15) * ldk + k0 + ((lane & 16) ? 8 : 0);
  v8bf c0 = *(const v8bf*)(p);
  v8bf c1 = *(const v8bf*)(p + 16);
  v16bf a;
#pragma unroll
  for (int e = 0; e < 8; ++e) { a[e] = c0[e]; a[e + 8] = c1[e]; }
  return a;
}
static __device__ __forceinline__ v16bf frag_b(const __bf16* tile, int ldk, int k0, int lane) {
  const __bf16* p = tile + (size_t)(lane & 15) * ldk + k0 + ((lane & 16) ? 8 : 0);
  v8bf c0 = *(const v8bf*)(p);
  v8bf c1 = *(const v8bf*)(p + 16);
  v16bf b;
#pragma unroll
  for (int e = 0; e < 8; ++e) { b[e] = c0[e]; b[e + 8] = c1[e]; }
  return b;
}

// ---- Async staging of one 16x32 f32 pos_bias tile into LDS (GV mode, ASYNCcnt) ----
// 4 instructions per wave; each moves 32 lanes x 16B = 512B (4 rows of 128B).
static __device__ __forceinline__ void async_pos_tile(unsigned ldsBase, const float* pbBase,
                                                      int j0, int lane) {
  unsigned rowsub = (unsigned)(lane >> 3);        // 0..3
  unsigned colb   = (unsigned)(lane & 7) * 16u;   // byte within 128B row
#pragma unroll
  for (int t = 0; t < 4; ++t) {
    unsigned r = (unsigned)t * 4u + rowsub;       // tile row 0..15
    unsigned lds = ldsBase + r * 128u + colb;
    unsigned long long ga = (unsigned long long)(uintptr_t)pbBase +
                            (unsigned long long)r * (N_ * 4u) + (unsigned)j0 * 4u + colb;
    asm volatile("global_load_async_to_lds_b128 %0, %1, off"
                 :: "v"(lds), "v"(ga) : "memory");
  }
}
static __device__ __forceinline__ void wait_async0() {
  asm volatile("s_wait_asynccnt 0x0" ::: "memory");
}

__global__ void k_cast_bf16(const float* __restrict__ in, __bf16* __restrict__ out, int n) {
  int i = blockIdx.x * blockDim.x + threadIdx.x;
  if (i < n) out[i] = (__bf16)in[i];
}

// ---- QKV projection: qkv[m,c] = sum_k x[m,k] * Wqkv[c,k]; scatter into q,k,vT + kv cache ----
__global__ __launch_bounds__(256) void k_qkv_gemm(
    const __bf16* __restrict__ X, const __bf16* __restrict__ W,
    __bf16* __restrict__ qb, __bf16* __restrict__ kb, __bf16* __restrict__ vT,
    float* __restrict__ kvout) {
  int lane = threadIdx.x & 31, wid = threadIdx.x >> 5;
  int c0 = (blockIdx.x * 8 + wid) * 16;
  int m0 = blockIdx.y * 16;
  const __bf16* Atile = X + (size_t)m0 * C_;
  const __bf16* Btile = W + (size_t)c0 * C_;
  v8f acc = {};
  for (int k0 = 0; k0 < C_; k0 += 32) {
    v16bf a = frag_a(Atile, C_, k0, lane);
    v16bf b = frag_b(Btile, C_, k0, lane);
    acc = __builtin_amdgcn_wmma_f32_16x16x32_bf16(false, a, false, b, (short)0, acc, false, false);
  }
  int c = c0 + (lane & 15);
  int h = c / 192, rem = c % 192, d = rem / 3, s = rem % 3;  // (h d qkv) fastest = qkv
#pragma unroll
  for (int r = 0; r < 8; ++r) {
    int m = m0 + r + ((lane & 16) ? 8 : 0);
    int bb = m >> 11, i = m & (N_ - 1);
    float v = acc[r];
    size_t bhnd = (((size_t)bb * H_ + h) * N_ + i) * D_ + d;
    if (s == 0) {
      qb[bhnd] = (__bf16)(v * 0.125f);                 // fold D^-0.5 = 1/8 into q
    } else if (s == 1) {
      kb[bhnd] = (__bf16)v;
      kvout[bhnd] = v;                                 // kv[0] = k
    } else {
      vT[(((size_t)bb * H_ + h) * D_ + d) * N_ + i] = (__bf16)v;
      kvout[(size_t)B_ * H_ * N_ * D_ + bhnd] = v;     // kv[1] = v
    }
  }
}

// ---- Pass 1: streaming softmax stats per (b,h,row) with pre talking-heads mix ----
__global__ __launch_bounds__(384) void k_attn_pass1(
    const __bf16* __restrict__ qb, const __bf16* __restrict__ kb,
    const float* __restrict__ pos_bias, const unsigned char* __restrict__ mask,
    const float* __restrict__ Wpre, const float* __restrict__ bpre,
    float* __restrict__ mOut, float* __restrict__ lOut) {
  __shared__ float sS[H_ * 16 * 33];                       // raw scores; reused for reduction
  __shared__ __align__(16) float posb[H_ * 16 * 32];       // async-staged pos_bias tiles
  int lane = threadIdx.x & 31, g = threadIdx.x >> 5;
  int b = blockIdx.x >> 7;                 // N_/16 == 128 tiles
  int i0 = (blockIdx.x & 127) * 16;
  int half8 = (lane & 16) ? 8 : 0;
  int cl = lane & 15;
  float wrow[H_];
#pragma unroll
  for (int h = 0; h < H_; ++h) wrow[h] = Wpre[g * H_ + h];
  float bias = bpre[g];
  const __bf16* qt = qb + (((size_t)b * H_ + g) * N_ + i0) * D_;
  const __bf16* kt = kb + ((size_t)b * H_ + g) * N_ * D_;
  v16bf a0 = frag_a(qt, D_, 0, lane);
  v16bf a1 = frag_a(qt, D_, 32, lane);
  const float* pbBase = pos_bias + (size_t)g * N_ * N_ + (size_t)i0 * N_;
  const unsigned char* mkBase = mask + (size_t)b * N_ * N_ + (size_t)i0 * N_;
  unsigned posOff = (unsigned)(uintptr_t)posb + (unsigned)g * 2048u;  // per-head 16*32*4B
  const float* posg = posb + g * 512;
  float m_run[8], l_run[8];
#pragma unroll
  for (int r = 0; r < 8; ++r) { m_run[r] = -FLT_MAX; l_run[r] = 0.f; }

  for (int j0 = 0; j0 < N_; j0 += 32) {
    async_pos_tile(posOff, pbBase, j0, lane);              // overlap with WMMA below
#pragma unroll
    for (int cg = 0; cg < 2; ++cg) {
      const __bf16* btile = kt + (size_t)(j0 + cg * 16) * D_;
      v16bf b0 = frag_b(btile, D_, 0, lane);
      v16bf b1 = frag_b(btile, D_, 32, lane);
      v8f s = {};
      s = __builtin_amdgcn_wmma_f32_16x16x32_bf16(false, a0, false, b0, (short)0, s, false, false);
      s = __builtin_amdgcn_wmma_f32_16x16x32_bf16(false, a1, false, b1, (short)0, s, false, false);
#pragma unroll
      for (int r = 0; r < 8; ++r)
        sS[g * 528 + (r + half8) * 33 + cg * 16 + cl] = s[r];
    }
    __syncthreads();
    wait_async0();                                          // pos tile resident in LDS
#pragma unroll
    for (int cg = 0; cg < 2; ++cg) {
#pragma unroll
      for (int r = 0; r < 8; ++r) {
        int M = r + half8, cc = cg * 16 + cl;
        float sv = bias;
#pragma unroll
        for (int h = 0; h < H_; ++h) sv += wrow[h] * sS[h * 528 + M * 33 + cc];
        sv += posg[M * 32 + cc];
        if (mkBase[(size_t)M * N_ + j0 + cc]) sv = -FLT_MAX;
        float m = m_run[r];
        if (sv > m) { l_run[r] = l_run[r] * __expf(m - sv) + 1.f; m_run[r] = sv; }
        else        { l_run[r] += __expf(sv - m); }
      }
    }
    __syncthreads();
  }
  // reduce across the 16 lanes holding each row (reuse sS as scratch)
  float* mred = sS;
  float* lred = sS + 3072;
#pragma unroll
  for (int r = 0; r < 8; ++r) {
    mred[(g * 16 + r + half8) * 16 + cl] = m_run[r];
    lred[(g * 16 + r + half8) * 16 + cl] = l_run[r];
  }
  __syncthreads();
  int t = threadIdx.x;
  if (t < H_ * 16) {
    float mm = -FLT_MAX;
#pragma unroll
    for (int q = 0; q < 16; ++q) mm = fmaxf(mm, mred[t * 16 + q]);
    float ll = 0.f;
#pragma unroll
    for (int q = 0; q < 16; ++q) ll += lred[t * 16 + q] * __expf(mred[t * 16 + q] - mm);
    int gg = t >> 4, M = t & 15;
    size_t idx = ((size_t)b * H_ + gg) * N_ + i0 + M;
    mOut[idx] = mm;
    lOut[idx] = ll;
  }
}

// ---- Pass 2: recompute scores, normalize, post talking-heads mix, P~ @ V ----
__global__ __launch_bounds__(384) void k_attn_pass2(
    const __bf16* __restrict__ qb, const __bf16* __restrict__ kb, const __bf16* __restrict__ vT,
    const float* __restrict__ pos_bias, const unsigned char* __restrict__ mask,
    const float* __restrict__ Wpre, const float* __restrict__ bpre,
    const float* __restrict__ Wpost, const float* __restrict__ bpost,
    const float* __restrict__ mIn, const float* __restrict__ lIn,
    __bf16* __restrict__ ao) {
  __shared__ float sS[H_ * 16 * 33];                      // raw scores (f32)
  __shared__ __bf16 pPb[H_ * 16 * 34];                    // normalized p (bf16, padded)
  __shared__ __align__(16) float posb[H_ * 16 * 32];      // async-staged pos_bias tiles
  __shared__ float lm[H_ * 16];
  __shared__ float lrcp[H_ * 16];
  int lane = threadIdx.x & 31, g = threadIdx.x >> 5;
  int b = blockIdx.x >> 7;
  int i0 = (blockIdx.x & 127) * 16;
  int half8 = (lane & 16) ? 8 : 0;
  int cl = lane & 15;
  float wpre[H_], wpost[H_];
#pragma unroll
  for (int h = 0; h < H_; ++h) { wpre[h] = Wpre[g * H_ + h]; wpost[h] = Wpost[g * H_ + h]; }
  float biasPre = bpre[g], biasPost = bpost[g];
  {
    int t = threadIdx.x;
    if (t < H_ * 16) {
      int gg = t >> 4, M = t & 15;
      size_t idx = ((size_t)b * H_ + gg) * N_ + i0 + M;
      lm[t] = mIn[idx];
      lrcp[t] = 1.f / lIn[idx];
    }
  }
  const __bf16* qt = qb + (((size_t)b * H_ + g) * N_ + i0) * D_;
  const __bf16* kt = kb + ((size_t)b * H_ + g) * N_ * D_;
  const __bf16* vth = vT + ((size_t)b * H_ + g) * D_ * N_;
  v16bf a0 = frag_a(qt, D_, 0, lane);
  v16bf a1 = frag_a(qt, D_, 32, lane);
  const float* pbBase = pos_bias + (size_t)g * N_ * N_ + (size_t)i0 * N_;
  const unsigned char* mkBase = mask + (size_t)b * N_ * N_ + (size_t)i0 * N_;
  unsigned posOff = (unsigned)(uintptr_t)posb + (unsigned)g * 2048u;
  const float* posg = posb + g * 512;
  v8f o[4] = {};
  __syncthreads();

  for (int j0 = 0; j0 < N_; j0 += 32) {
    async_pos_tile(posOff, pbBase, j0, lane);
#pragma unroll
    for (int cg = 0; cg < 2; ++cg) {
      const __bf16* btile = kt + (size_t)(j0 + cg * 16) * D_;
      v16bf b0 = frag_b(btile, D_, 0, lane);
      v16bf b1 = frag_b(btile, D_, 32, lane);
      v8f s = {};
      s = __builtin_amdgcn_wmma_f32_16x16x32_bf16(false, a0, false, b0, (short)0, s, false, false);
      s = __builtin_amdgcn_wmma_f32_16x16x32_bf16(false, a1, false, b1, (short)0, s, false, false);
#pragma unroll
      for (int r = 0; r < 8; ++r)
        sS[g * 528 + (r + half8) * 33 + cg * 16 + cl] = s[r];
    }
    __syncthreads();
    wait_async0();
    // pre-mix + pos + mask -> normalized p (pass-1 stats), into pPb (bf16)
#pragma unroll
    for (int cg = 0; cg < 2; ++cg) {
#pragma unroll
      for (int r = 0; r < 8; ++r) {
        int M = r + half8, cc = cg * 16 + cl;
        float sv = biasPre;
#pragma unroll
        for (int h = 0; h < H_; ++h) sv += wpre[h] * sS[h * 528 + M * 33 + cc];
        sv += posg[M * 32 + cc];
        if (mkBase[(size_t)M * N_ + j0 + cc]) sv = -FLT_MAX;
        pPb[g * 544 + M * 34 + cc] = (__bf16)(__expf(sv - lm[g * 16 + M]) * lrcp[g * 16 + M]);
      }
    }
    __syncthreads();
    // post-mix on the fly while building A-fragment of P~ (16x32 bf16)
    v16bf af;
    {
      int row = lane & 15;
      int koff = (lane & 16) ? 8 : 0;
#pragma unroll
      for (int e = 0; e < 16; ++e) {
        int k = (e < 8) ? (koff + e) : (16 + koff + e - 8);
        float pv = biasPost;
#pragma unroll
        for (int h = 0; h < H_; ++h) pv += wpost[h] * (float)pPb[h * 544 + row * 34 + k];
        af[e] = (__bf16)pv;
      }
    }
#pragma unroll
    for (int dg = 0; dg < 4; ++dg) {
      v16bf vb = frag_b(vth + (size_t)(dg * 16) * N_, N_, j0, lane);
      o[dg] = __builtin_amdgcn_wmma_f32_16x16x32_bf16(false, af, false, vb, (short)0, o[dg], false, false);
    }
    __syncthreads();
  }
  // store attention output in [B, N, H*D] bf16 for the final projection
#pragma unroll
  for (int dg = 0; dg < 4; ++dg)
#pragma unroll
    for (int r = 0; r < 8; ++r)
      ao[((size_t)b * N_ + i0 + r + half8) * HD_ + g * 64 + dg * 16 + cl] = (__bf16)o[dg][r];
}

// ---- Output projection: out[m,c] = sum_k ao[m,k] * Wout[c,k] (fp32 out) ----
__global__ __launch_bounds__(256) void k_out_gemm(
    const __bf16* __restrict__ A, const __bf16* __restrict__ W, float* __restrict__ out) {
  int lane = threadIdx.x & 31, wid = threadIdx.x >> 5;
  int c0 = (blockIdx.x * 8 + wid) * 16;
  int m0 = blockIdx.y * 16;
  const __bf16* Atile = A + (size_t)m0 * HD_;
  const __bf16* Btile = W + (size_t)c0 * HD_;
  v8f acc = {};
  for (int k0 = 0; k0 < HD_; k0 += 32) {
    v16bf a = frag_a(Atile, HD_, k0, lane);
    v16bf b = frag_b(Btile, HD_, k0, lane);
    acc = __builtin_amdgcn_wmma_f32_16x16x32_bf16(false, a, false, b, (short)0, acc, false, false);
  }
#pragma unroll
  for (int r = 0; r < 8; ++r) {
    int m = m0 + r + ((lane & 16) ? 8 : 0);
    out[(size_t)m * C_ + c0 + (lane & 15)] = acc[r];
  }
}

extern "C" void kernel_launch(void* const* d_in, const int* in_sizes, int n_in,
                              void* d_out, int out_size, void* d_ws, size_t ws_size,
                              hipStream_t stream) {
  (void)in_sizes; (void)n_in; (void)out_size; (void)ws_size;
  const float* x    = (const float*)d_in[0];
  const float* pos  = (const float*)d_in[1];
  const unsigned char* mask = (const unsigned char*)d_in[2];
  const float* Wqkv = (const float*)d_in[3];
  const float* Wout = (const float*)d_in[4];
  const float* Wpre = (const float*)d_in[5];
  const float* bpre = (const float*)d_in[6];
  const float* Wpos = (const float*)d_in[7];
  const float* bpos = (const float*)d_in[8];
  float* out   = (float*)d_out;
  float* kvout = out + (size_t)B_ * N_ * C_;    // kv: [2,B,H,N,D] after out: [B,N,C]

  char* ws = (char*)d_ws;
  size_t off = 0;
  auto alloc = [&](size_t bytes) { char* p = ws + off; off = (off + bytes + 255) & ~(size_t)255; return p; };
  __bf16* xb  = (__bf16*)alloc((size_t)B_ * N_ * C_ * 2);
  __bf16* wqb = (__bf16*)alloc((size_t)QKV_ * C_ * 2);
  __bf16* wob = (__bf16*)alloc((size_t)C_ * HD_ * 2);
  __bf16* qb  = (__bf16*)alloc((size_t)B_ * H_ * N_ * D_ * 2);
  __bf16* kb  = (__bf16*)alloc((size_t)B_ * H_ * N_ * D_ * 2);
  __bf16* vT  = (__bf16*)alloc((size_t)B_ * H_ * N_ * D_ * 2);
  __bf16* ao  = (__bf16*)alloc((size_t)B_ * N_ * HD_ * 2);
  float* mbuf = (float*)alloc((size_t)B_ * H_ * N_ * 4);
  float* lbuf = (float*)alloc((size_t)B_ * H_ * N_ * 4);

  int n1 = B_ * N_ * C_;
  k_cast_bf16<<<(n1 + 255) / 256, 256, 0, stream>>>(x, xb, n1);
  int n2 = QKV_ * C_;
  k_cast_bf16<<<(n2 + 255) / 256, 256, 0, stream>>>(Wqkv, wqb, n2);
  int n3 = C_ * HD_;
  k_cast_bf16<<<(n3 + 255) / 256, 256, 0, stream>>>(Wout, wob, n3);

  k_qkv_gemm<<<dim3(QKV_ / 128, (B_ * N_) / 16), 256, 0, stream>>>(xb, wqb, qb, kb, vT, kvout);
  k_attn_pass1<<<B_ * (N_ / 16), 384, 0, stream>>>(qb, kb, pos, mask, Wpre, bpre, mbuf, lbuf);
  k_attn_pass2<<<B_ * (N_ / 16), 384, 0, stream>>>(qb, kb, vT, pos, mask, Wpre, bpre, Wpos, bpos,
                                                   mbuf, lbuf, ao);
  k_out_gemm<<<dim3(HD_ / 128, (B_ * N_) / 16), 256, 0, stream>>>(ao, wob, out);
}
